// RoutedLoRAConv1D_16707422781874
// MI455X (gfx1250) — matrix operand order repaired and checked by reading, not compiled, not run on hardware
//
#include <hip/hip_runtime.h>

// Routed LoRA Conv1D for gfx1250 (MI455X).
// Strategy: one-shot fp32->bf16 conversion passes into workspace (x, W^T,
// lora_a^T, lora_b^T all bf16, ~43MB, L2-resident), then bf16 WMMA GEMM with
// f32 accumulate. Main GEMM double-buffers LDS tiles via CDNA5 async
// global->LDS copies (ASYNCcnt) when available. LoRA delta fused as one
// zero-padded-K WMMA per output tile; bias fused in the store.

typedef __attribute__((ext_vector_type(16))) __bf16 v16bf;
typedef __attribute__((ext_vector_type(8)))  __bf16 v8bf;
typedef __attribute__((ext_vector_type(8)))  float  v8f;
typedef __attribute__((ext_vector_type(4)))  int    v4i;

constexpr int IN_DIM  = 1024;
constexpr int OUT_DIM = 4096;
constexpr int RANK    = 16;
constexpr int M_TOT   = 32 * 512;       // B * S
constexpr float SCALING = 16.0f / 16.0f;

// ---- CDNA5 async global->LDS copy (guarded; sync-copy fallback) ------------
#if defined(__has_builtin)
#if __has_builtin(__builtin_amdgcn_global_load_async_to_lds_b128)
#define HAVE_ASYNC_LDS 1
#endif
#endif

typedef __attribute__((address_space(1))) v4i gv4i;   // global int4
typedef __attribute__((address_space(3))) v4i lv4i;   // LDS int4

static __device__ inline void cp16(const __bf16* g, __bf16* l) {
#if defined(HAVE_ASYNC_LDS)
  __builtin_amdgcn_global_load_async_to_lds_b128((gv4i*)g, (lv4i*)l, 0, 0);
#else
  *(v8bf*)l = *(const v8bf*)g;
#endif
}

static __device__ inline void wait_async4() {
#if defined(HAVE_ASYNC_LDS)
#if __has_builtin(__builtin_amdgcn_s_wait_asynccnt)
  __builtin_amdgcn_s_wait_asynccnt(4);
#else
  asm volatile("s_wait_asynccnt 0x4" ::: "memory");
#endif
#endif
}
static __device__ inline void wait_async0() {
#if defined(HAVE_ASYNC_LDS)
#if __has_builtin(__builtin_amdgcn_s_wait_asynccnt)
  __builtin_amdgcn_s_wait_asynccnt(0);
#else
  asm volatile("s_wait_asynccnt 0x0" ::: "memory");
#endif
#endif
}

// ---- fragment helpers ------------------------------------------------------
// v16bf from two 16B chunks (elems 0..7 from p0, 8..15 from p1)
static __device__ inline v16bf ld2(const __bf16* p0, const __bf16* p1) {
  v8bf lo = *(const v8bf*)p0;
  v8bf hi = *(const v8bf*)p1;
  return __builtin_shufflevector(lo, hi, 0,1,2,3,4,5,6,7,8,9,10,11,12,13,14,15);
}

static __device__ inline v8bf zero8() {
  v8bf z;
  #pragma unroll
  for (int i = 0; i < 8; ++i) z[i] = (__bf16)0.0f;
  return z;
}

static __device__ inline v8f wmma_bf16(v16bf a, v16bf b, v8f c) {
  // (neg_a, A, neg_b, B, c_mod, C, reuse_a, reuse_b)
  return __builtin_amdgcn_wmma_f32_16x16x32_bf16(false, a, false, b, (short)0, c, false, false);
}

// ---------------------------------------------------------------------------
// One-shot conversion / transpose passes (memory-bound, ~4us total)
// ---------------------------------------------------------------------------
__global__ __launch_bounds__(256) void k_cvt_x(const float* __restrict__ x,
                                               __bf16* __restrict__ xb) {
  size_t base = ((size_t)blockIdx.x * 256 + threadIdx.x) * 8;
  float4 v0 = *(const float4*)(x + base);
  float4 v1 = *(const float4*)(x + base + 4);
  __bf16* d = xb + base;
  d[0] = (__bf16)v0.x; d[1] = (__bf16)v0.y; d[2] = (__bf16)v0.z; d[3] = (__bf16)v0.w;
  d[4] = (__bf16)v1.x; d[5] = (__bf16)v1.y; d[6] = (__bf16)v1.z; d[7] = (__bf16)v1.w;
}

// W[K,N] fp32 -> WT[N,K] bf16
__global__ __launch_bounds__(256) void k_tr_w(const float* __restrict__ W,
                                              __bf16* __restrict__ WT) {
  int f = blockIdx.x * 256 + threadIdx.x;     // 0 .. (1024*4096/4 - 1)
  int k = f >> 10;
  int n = (f & 1023) * 4;
  float4 v = *(const float4*)(W + (size_t)k * OUT_DIM + n);
  WT[(size_t)(n + 0) * IN_DIM + k] = (__bf16)v.x;
  WT[(size_t)(n + 1) * IN_DIM + k] = (__bf16)v.y;
  WT[(size_t)(n + 2) * IN_DIM + k] = (__bf16)v.z;
  WT[(size_t)(n + 3) * IN_DIM + k] = (__bf16)v.w;
}

// lora_a[a][K=1024][R=16] fp32 -> laT[a][R][K] bf16
__global__ __launch_bounds__(256) void k_tr_la(const float* __restrict__ la,
                                               __bf16* __restrict__ laT) {
  int a = blockIdx.x;
  const float* src = la + (size_t)a * (IN_DIM * RANK);
  __bf16* dst = laT + (size_t)a * (IN_DIM * RANK);
  for (int it = 0; it < (IN_DIM * RANK) / 256; ++it) {
    int e = it * 256 + threadIdx.x;
    int k = e >> 4, n = e & 15;
    dst[n * IN_DIM + k] = (__bf16)src[e];
  }
}

// lora_b[a][R=16][N=4096] fp32 -> lbT[a][N][R] bf16
__global__ __launch_bounds__(256) void k_tr_lb(const float* __restrict__ lb,
                                               __bf16* __restrict__ lbT) {
  int a = blockIdx.x;
  const float* src = lb + (size_t)a * (RANK * OUT_DIM);
  __bf16* dst = lbT + (size_t)a * (RANK * OUT_DIM);
  for (int it = 0; it < (RANK * OUT_DIM) / (256 * 4); ++it) {
    int f = it * 256 + threadIdx.x;           // 0..16383
    int k = f >> 10;
    int n = (f & 1023) * 4;
    float4 v = *(const float4*)(src + (size_t)k * OUT_DIM + n);
    dst[(n + 0) * RANK + k] = (__bf16)v.x;
    dst[(n + 1) * RANK + k] = (__bf16)v.y;
    dst[(n + 2) * RANK + k] = (__bf16)v.z;
    dst[(n + 3) * RANK + k] = (__bf16)v.w;
  }
}

// ---------------------------------------------------------------------------
// Kernel: low_rank[M,R] = (x @ lora_a[id]) * SCALING   (bf16 out, no LDS)
// One WG per 128-row slab; slab lies in one sample -> single adapter id.
// ---------------------------------------------------------------------------
__global__ __launch_bounds__(256) void k_lowrank(
    const __bf16* __restrict__ xb, const int* __restrict__ ids,
    const __bf16* __restrict__ laT, __bf16* __restrict__ lrB) {
  const int tid = threadIdx.x;
  const int m0  = blockIdx.x * 128;
  const int id  = ids[m0 >> 9];
  const __bf16* la = laT + (size_t)id * (IN_DIM * RANK);

  const int w  = tid >> 5, lane = tid & 31;
  const int lm = lane & 15, kh = lane >> 4, koff = kh * 8;
  const int mrow = m0 + w * 16 + lm;

  const __bf16* xr = xb + (size_t)mrow * IN_DIM;   // A row for this lane
  const __bf16* br = la + (size_t)lm * IN_DIM;     // A^T row = column lm

  v8f acc;
  #pragma unroll
  for (int i = 0; i < 8; ++i) acc[i] = 0.0f;

  for (int kb = 0; kb < IN_DIM; kb += 32) {
    v16bf a = ld2(xr + kb + koff, xr + kb + 16 + koff);
    v16bf b = ld2(br + kb + koff, br + kb + 16 + koff);
    acc = wmma_bf16(a, b, acc);
  }

  #pragma unroll
  for (int r = 0; r < 8; ++r)
    lrB[(size_t)(m0 + w * 16 + r + 8 * kh) * RANK + lm] = (__bf16)(acc[r] * SCALING);
}

// ---------------------------------------------------------------------------
// Main: out = x @ W + b + low_rank @ lora_b[id]
// 128x128 tile/WG; 8 waves as 4(M) x 2(N); wave = 2x4 WMMA tiles.
// Async double-buffered LDS staging; LoRA epilogue reads global bf16 directly.
// ---------------------------------------------------------------------------
__global__ __launch_bounds__(256) void k_main(
    const __bf16* __restrict__ xb, const int* __restrict__ ids,
    const __bf16* __restrict__ WT, const float* __restrict__ bias,
    const __bf16* __restrict__ lbT, const __bf16* __restrict__ lrB,
    float* __restrict__ out) {
  constexpr int AST = 40;                    // 32 + 8 pad: bank-spread, 16B rows
  __shared__ __bf16 As[2][128 * AST];        // x tile (M x 32), 2 buffers
  __shared__ __bf16 Bs[2][128 * AST];        // W^T tile (N x 32), 2 buffers

  const int tid = threadIdx.x;
  const int n0  = blockIdx.x * 128;
  const int m0  = blockIdx.y * 128;
  const int id  = ids[m0 >> 9];

  const __bf16* xA = xb + (size_t)m0 * IN_DIM;
  const __bf16* wB = WT + (size_t)n0 * IN_DIM;

  const int w  = tid >> 5, lane = tid & 31;
  const int wm = w >> 1, wn = w & 1;
  const int lm = lane & 15, kh = lane >> 4, koff = kh * 8;

  // stage tile at K-offset kb into buffer buf (4 async 16B chunks per thread)
  auto issue = [&](int kb, int buf) {
    #pragma unroll
    for (int q = 0; q < 2; ++q) {
      int f = q * 256 + tid;                 // 0..511
      int r = f >> 2, c = (f & 3) * 8;
      cp16(xA + (size_t)r * IN_DIM + kb + c, &As[buf][r * AST + c]);
    }
    #pragma unroll
    for (int q = 0; q < 2; ++q) {
      int f = q * 256 + tid;
      int n = f >> 2, c = (f & 3) * 8;
      cp16(wB + (size_t)n * IN_DIM + kb + c, &Bs[buf][n * AST + c]);
    }
  };

  v8f acc[2][4];
  #pragma unroll
  for (int i = 0; i < 2; ++i)
    #pragma unroll
    for (int j = 0; j < 4; ++j)
      #pragma unroll
      for (int r = 0; r < 8; ++r) acc[i][j][r] = 0.0f;

  constexpr int NK = IN_DIM / 32;            // 32 K-steps
  issue(0, 0);
  issue(32, 1);

  for (int it = 0; it < NK; ++it) {
    const int cur = it & 1;
    if (it < NK - 1) wait_async4(); else wait_async0();
    __syncthreads();                         // tile `it` visible to all waves

    v16bf af[2];
    #pragma unroll
    for (int i = 0; i < 2; ++i) {
      const __bf16* p = &As[cur][(wm * 32 + i * 16 + lm) * AST];
      af[i] = ld2(p + koff, p + 16 + koff);
    }
    v16bf bfr[4];
    #pragma unroll
    for (int j = 0; j < 4; ++j) {
      const __bf16* p = &Bs[cur][(wn * 64 + j * 16 + lm) * AST + koff];
      bfr[j] = ld2(p, p + 16);
    }
    #pragma unroll
    for (int i = 0; i < 2; ++i)
      #pragma unroll
      for (int j = 0; j < 4; ++j)
        acc[i][j] = wmma_bf16(af[i], bfr[j], acc[i][j]);

    __syncthreads();                         // all waves done reading buf cur
    if (it + 2 < NK) issue((it + 2) * 32, cur);
  }

  // LoRA epilogue: K=16 zero-padded to 32; fragments straight from global bf16
  v8bf z8 = zero8();
  v16bf alr[2];
  #pragma unroll
  for (int i = 0; i < 2; ++i) {
    const __bf16* p = lrB + (size_t)(m0 + wm * 32 + i * 16 + lm) * RANK;
    v8bf lo = *(const v8bf*)(p + koff);
    alr[i] = __builtin_shufflevector(lo, z8, 0,1,2,3,4,5,6,7,8,9,10,11,12,13,14,15);
  }
  v16bf bep[4];
  #pragma unroll
  for (int j = 0; j < 4; ++j) {
    const __bf16* p = lbT + (size_t)id * (RANK * OUT_DIM)
                    + (size_t)(n0 + wn * 64 + j * 16 + lm) * RANK;
    v8bf lo = *(const v8bf*)(p + koff);
    v8bf hi = z8;                            // K >= 16 rows are zero
    bep[j] = __builtin_shufflevector(lo, hi, 0,1,2,3,4,5,6,7,8,9,10,11,12,13,14,15);
  }
  #pragma unroll
  for (int i = 0; i < 2; ++i)
    #pragma unroll
    for (int j = 0; j < 4; ++j)
      acc[i][j] = wmma_bf16(alr[i], bep[j], acc[i][j]);

  // bias + store
  #pragma unroll
  for (int j = 0; j < 4; ++j) {
    int n = n0 + wn * 64 + j * 16 + lm;
    float bv = bias[n];
    #pragma unroll
    for (int i = 0; i < 2; ++i) {
      int rbase = m0 + wm * 32 + i * 16 + kh * 8;
      #pragma unroll
      for (int r = 0; r < 8; ++r)
        out[(size_t)(rbase + r) * OUT_DIM + n] = acc[i][j][r] + bv;
    }
  }
}

extern "C" void kernel_launch(void* const* d_in, const int* in_sizes, int n_in,
                              void* d_out, int out_size, void* d_ws, size_t ws_size,
                              hipStream_t stream) {
  (void)in_sizes; (void)n_in; (void)out_size; (void)ws_size;
  const float* x      = (const float*)d_in[0];
  const int*   ids    = (const int*)  d_in[1];
  const float* W      = (const float*)d_in[2];
  const float* bias   = (const float*)d_in[3];
  const float* lora_a = (const float*)d_in[4];
  const float* lora_b = (const float*)d_in[5];
  float* out = (float*)d_out;

  // workspace layout (bf16), total ~43 MB
  char* ws = (char*)d_ws;
  __bf16* xb  = (__bf16*)(ws);                         // 16M elems  = 32 MB
  __bf16* WT  = (__bf16*)(ws + 33554432);              // 4M elems   =  8 MB
  __bf16* laT = (__bf16*)(ws + 41943040);              // 256K elems = 512 KB
  __bf16* lbT = (__bf16*)(ws + 42467328);              // 1M elems   =  2 MB
  __bf16* lrB = (__bf16*)(ws + 44564480);              // 256K elems = 512 KB

  k_cvt_x<<<dim3((M_TOT * IN_DIM) / (256 * 8)), 256, 0, stream>>>(x, xb);
  k_tr_w <<<dim3((IN_DIM * OUT_DIM) / (256 * 4)), 256, 0, stream>>>(W, WT);
  k_tr_la<<<dim3(16), 256, 0, stream>>>(lora_a, laT);
  k_tr_lb<<<dim3(16), 256, 0, stream>>>(lora_b, lbT);
  k_lowrank<<<dim3(M_TOT / 128), 256, 0, stream>>>(xb, ids, laT, lrB);
  k_main<<<dim3(OUT_DIM / 128, M_TOT / 128), 256, 0, stream>>>(
      xb, ids, WT, bias, lbT, lrB, out);
}
